// LM_13022340842314
// MI455X (gfx1250) — matrix operand order repaired and checked by reading, not compiled, not run on hardware
//
#include <hip/hip_runtime.h>
#include <hip/hip_bf16.h>
#include <math.h>

// Problem dims (fixed by the reference)
static constexpr int B_ = 4, T_ = 256, S_ = 512, D_ = 512, V_ = 50257;
static constexpr int BT = B_ * T_;   // 1024
static constexpr int BS = B_ * S_;   // 2048

typedef __bf16 bhalf;
typedef __attribute__((ext_vector_type(16))) __bf16 v16bf;
typedef __attribute__((ext_vector_type(8)))  float  v8f;

typedef unsigned int u32x4 __attribute__((ext_vector_type(4)));
typedef int          i32x8 __attribute__((ext_vector_type(8)));
typedef int          i32x4 __attribute__((ext_vector_type(4)));

// ---------------------------------------------------------------------------
// WMMA fragment loaders, per CDNA5 ISA 7.12.2 (wave32):
//  A 16x32 bf16:  lane L holds row M=L&15; K chunks [k0..k0+7] and [k0+16..k0+23]
//                 with k0 = (L>=16)?8:0  -> two aligned 16B ds_load_b128
//  B 32x16 bf16 (stored transposed Bt[n][k], k contiguous): lane L holds col
//                 N=L&15; K chunks [k0..k0+7] and [k0+16..k0+23], k0 = (L>=16)?8:0
//  C/D f32 16x16: lane L col N=L&15, VGPR r row M = r + ((L>=16)?8:0)
// ---------------------------------------------------------------------------
__device__ __forceinline__ v16bf frag_a(const bhalf* base, int ldm, int lane) {
  const bhalf* p = base + (lane & 15) * ldm + ((lane & 16) ? 8 : 0);
  v16bf a;
  ((float4*)&a)[0] = *(const float4*)(p);
  ((float4*)&a)[1] = *(const float4*)(p + 16);
  return a;
}
__device__ __forceinline__ v16bf frag_b(const bhalf* bt, int ldn, int lane) {
  const bhalf* p = bt + (lane & 15) * ldn + ((lane & 16) ? 8 : 0);
  v16bf b;
  ((float4*)&b)[0] = *(const float4*)(p);
  ((float4*)&b)[1] = *(const float4*)(p + 16);
  return b;
}
__device__ __forceinline__ v8f wmma_bf16(v16bf a, v16bf b, v8f c) {
  return __builtin_amdgcn_wmma_f32_16x16x32_bf16(false, a, false, b, (short)0, c,
                                                 false, false);
}

// ---------------------------------------------------------------------------
// Tensor Data Mover: 2D tile load Global->LDS (D# built per ISA ch.8).
//   tile: tile_x elements (2B each) x tile_y rows, row pitch stride_x elems.
//   pad_enable inserts (pad_amount+1) DWORDs into the LDS stream after every
//   2^pad_interval * 8 bytes -> gives the LDS tile a padded row pitch.
// This toolchain declares the 6-arg builtin:
//   (u32x4 g0, i32x8 g1, i32x4 g2, i32x4 g3, i32x8, i32 cpol)
// ---------------------------------------------------------------------------
#if __has_builtin(__builtin_amdgcn_tensor_load_to_lds)
#define HAVE_TDM 1
__device__ __forceinline__ void tdm_load_2d_bf16(unsigned int lds_off, unsigned long long gaddr,
                                                 unsigned int tile_x, unsigned int tile_y,
                                                 unsigned long long stride_x,
                                                 unsigned int pad_interval,
                                                 unsigned int pad_amount) {
  const unsigned int tdim0 = 1u << 20, tdim1 = 1u << 20;  // never OOB here
  u32x4 g0;
  g0[0] = 1u;                                             // count=1, user D#
  g0[1] = lds_off;                                        // lds_addr (bytes)
  g0[2] = (unsigned int)(gaddr & 0xffffffffull);          // global_addr[31:0]
  g0[3] = (unsigned int)((gaddr >> 32) & 0x01ffffffull)   // global_addr[56:32]
          | (2u << 30);                                   // type=2 ("image")
  i32x8 g1;
  g1[0] = (int)((1u << 16)                // data_size=1 (2 bytes)
                | (1u << 20)              // pad_enable
                | (pad_interval << 22)    // pad after 2^pi * 8 bytes
                | (pad_amount << 25));    // pad (pa+1) DWORDs
  g1[1] = (int)((tdim0 & 0xffffu) << 16);                 // tensor_dim0[15:0]
  g1[2] = (int)((tdim0 >> 16) | ((tdim1 & 0xffffu) << 16));
  g1[3] = (int)((tdim1 >> 16) | (tile_x << 16));          // tile_dim0
  g1[4] = (int)(tile_y & 0xffffu);                        // tile_dim1, tile_dim2=0
  g1[5] = (int)(unsigned int)(stride_x & 0xffffffffull);  // tensor_dim0_stride
  g1[6] = (int)(unsigned int)((stride_x >> 32) & 0xffffull);
  g1[7] = 0;
  i32x4 z4 = {0, 0, 0, 0};
  i32x8 z8 = {0, 0, 0, 0, 0, 0, 0, 0};
  __builtin_amdgcn_tensor_load_to_lds(g0, g1, z4, z4, z8, 0);
}
#else
#define HAVE_TDM 0
#endif

// ---------------------------------------------------------------------------
// fp32 -> bf16 cast
// ---------------------------------------------------------------------------
__global__ void k_cast(const float* __restrict__ src, bhalf* __restrict__ dst, int n) {
  for (int i = blockIdx.x * 256 + threadIdx.x; i < n; i += gridDim.x * 256)
    dst[i] = (bhalf)src[i];
}

// ---------------------------------------------------------------------------
// Generic bf16 WMMA GEMM: out = act(A[MxK] @ B[KxN] + bias)
// block = 256 threads (8 waves), tile 64M x 128N, K-step 32, double-buffered
// LDS pipeline (global loads for step t+1 issued before computing step t).
// Wave w -> rows (w&3)*16, cols (w>>2)*64 (1 A-frag x 4 B-frags = 4 WMMA/step).
// ---------------------------------------------------------------------------
template <int ACT>  // 0 = none, 1 = tanh
__global__ void k_gemm(const bhalf* __restrict__ A, const bhalf* __restrict__ Bm,
                       const float* __restrict__ bias, float* __restrict__ outF,
                       bhalf* __restrict__ outB, int M, int N, int K) {
  __shared__ bhalf As[2][64 * 40];    // 2 x  5 KB
  __shared__ bhalf Bt[2][128 * 40];   // 2 x 10 KB
  const int tid = threadIdx.x, lane = tid & 31, w = tid >> 5;
  const int m0 = blockIdx.y * 64, n0 = blockIdx.x * 128;
  const int mw = (w & 3) * 16, nw = (w >> 2) * 64;
  v8f acc[4] = {{}, {}, {}, {}};

  // per-thread staging registers
  float4 areg;
  bhalf breg[2][8];
  const int am = tid >> 2, aseg = (tid & 3) * 8;      // A: 64 rows x 4 chunks
  const int kr0 = tid >> 4, ns0 = (tid & 15) * 8;     // B chunk 0
  const int kr1 = (tid + 256) >> 4, ns1 = ns0;        // B chunk 1 (cc = tid+256)

  auto gload = [&](int kk) {
    int gm = m0 + am;
    areg = make_float4(0.f, 0.f, 0.f, 0.f);
    if (gm < M) areg = *(const float4*)(A + (size_t)gm * K + kk + aseg);
    {
      int gk = kk + kr0, gn = n0 + ns0;
      if (gn + 7 < N) *(float4*)breg[0] = *(const float4*)(Bm + (size_t)gk * N + gn);
      else for (int e = 0; e < 8; ++e)
        breg[0][e] = (gn + e < N) ? Bm[(size_t)gk * N + gn + e] : (bhalf)0.f;
    }
    {
      int gk = kk + kr1, gn = n0 + ns1;
      if (gn + 7 < N) *(float4*)breg[1] = *(const float4*)(Bm + (size_t)gk * N + gn);
      else for (int e = 0; e < 8; ++e)
        breg[1][e] = (gn + e < N) ? Bm[(size_t)gk * N + gn + e] : (bhalf)0.f;
    }
  };
  auto lstore = [&](int buf) {
    *(float4*)(As[buf] + am * 40 + aseg) = areg;
    for (int e = 0; e < 8; ++e) Bt[buf][(ns0 + e) * 40 + kr0] = breg[0][e];
    for (int e = 0; e < 8; ++e) Bt[buf][(ns1 + e) * 40 + kr1] = breg[1][e];
  };

  const int nsteps = K / 32;
  gload(0);
  lstore(0);
  __syncthreads();
  for (int t = 0; t < nsteps; ++t) {
    const int buf = t & 1;
    if (t + 1 < nsteps) gload((t + 1) * 32);   // overlap next global fetch
    v16bf a = frag_a(As[buf] + mw * 40, 40, lane);
    for (int f = 0; f < 4; ++f) {
      v16bf b = frag_b(Bt[buf] + (nw + f * 16) * 40, 40, lane);
      acc[f] = wmma_bf16(a, b, acc[f]);
    }
    if (t + 1 < nsteps) lstore(buf ^ 1);
    __syncthreads();
  }

  const int mbase = m0 + mw + ((lane & 16) ? 8 : 0);
  const int nc = lane & 15;
  for (int f = 0; f < 4; ++f) {
    int gn = n0 + nw + f * 16 + nc;
    if (gn >= N) continue;
    float bv = bias ? bias[gn] : 0.f;
    for (int r = 0; r < 8; ++r) {
      int gm = mbase + r;
      if (gm >= M) continue;
      float v = acc[f][r] + bv;
      if (ACT == 1) v = tanhf(v);
      if (outF) outF[(size_t)gm * N + gn] = v;
      if (outB) outB[(size_t)gm * N + gn] = (bhalf)v;
    }
  }
}

// ---------------------------------------------------------------------------
// Fused attention: one block per (b, 16-row t-tile).
// Phase 1: scores[16 x 512] = dec16 @ proj^T (WMMA, K=D), masked -> LDS fp32
//          (B tile staged by the Tensor Data Mover: 512x32 bf16 slice with
//           hardware padding to the 40-half LDS pitch)
// Phase 2: row softmax (fp32) -> attn_scores (global fp32) + P (LDS bf16)
// Phase 3: attn_out[16 x 512] = P @ mem (WMMA, K=S) -> bf16
// ---------------------------------------------------------------------------
__global__ void k_attn(const bhalf* __restrict__ decb, const bhalf* __restrict__ memb,
                       const bhalf* __restrict__ proj, const int* __restrict__ seqlen,
                       float* __restrict__ ascore, bhalf* __restrict__ attnb) {
  __shared__ float sc[16 * 520];       // 33.3 KB scores
  __shared__ bhalf decA[16 * 512];     // 16 KB   A tile (full K)
  __shared__ bhalf bt[512 * 40];       // 40 KB   staged B (padded-pitch layout)
  __shared__ bhalf pb[16 * 520];       // 16.6 KB softmax probs (bf16 A for phase 3)
  const int tid = threadIdx.x, lane = tid & 31, w = tid >> 5;
  const int b = blockIdx.x >> 4, t0 = (blockIdx.x & 15) * 16;
  const int L = seqlen[b];

  for (int idx = tid * 8; idx < 16 * 512; idx += 256 * 8) {
    int row = idx >> 9, col = idx & 511;
    *(float4*)(decA + idx) =
        *(const float4*)(decb + (size_t)(b * T_ + t0 + row) * D_ + col);
  }

  // ---- phase 1: scores = dec @ proj^T ----
  const int sw = w * 64;
  v8f acc[4] = {{}, {}, {}, {}};
  for (int dstep = 0; dstep < 16; ++dstep) {
    const int d0 = dstep * 32;
    __syncthreads();
#if HAVE_TDM
    // TDM: proj[b][0..511][d0..d0+31] -> bt with 40-half pitch.
    // row = 32 halves = 64B: pad_interval=3 (after 8x8B), pad_amount=3 (4 DW=16B)
    if (w == 0) {
      tdm_load_2d_bf16((unsigned int)(uintptr_t)bt,
                       (unsigned long long)(uintptr_t)(proj + (size_t)b * S_ * D_) +
                           (unsigned long long)d0 * 2ull,
                       /*tile_x=*/32, /*tile_y=*/512, /*stride_x=*/(unsigned long long)D_,
                       /*pad_interval=*/3, /*pad_amount=*/3);
      __builtin_amdgcn_s_wait_tensorcnt(0);
    }
#else
    for (int cc = tid; cc < 2048; cc += 256) {
      int s = cc >> 2, seg = (cc & 3) * 8;
      *(float4*)(bt + s * 40 + seg) =
          *(const float4*)(proj + (size_t)(b * S_ + s) * D_ + d0 + seg);
    }
#endif
    __syncthreads();
    v16bf a = frag_a(decA + d0, 512, lane);
    for (int f = 0; f < 4; ++f) {
      v16bf bb = frag_b(bt + (sw + f * 16) * 40, 40, lane);
      acc[f] = wmma_bf16(a, bb, acc[f]);
    }
  }
  {  // store masked scores
    int roff = (lane & 16) ? 8 : 0, nc = lane & 15;
    for (int f = 0; f < 4; ++f) {
      int scol = sw + f * 16 + nc;
      for (int r = 0; r < 8; ++r)
        sc[(roff + r) * 520 + scol] = (scol < L) ? acc[f][r] : -INFINITY;
    }
  }
  __syncthreads();

  // ---- phase 2: softmax over S (16 lanes per row, rows within one wave) ----
  {
    int row = tid >> 4, sub = tid & 15;
    float m = -INFINITY;
    for (int s = sub; s < 512; s += 16) m = fmaxf(m, sc[row * 520 + s]);
    for (int off = 8; off >= 1; off >>= 1) m = fmaxf(m, __shfl_xor(m, off, 16));
    float sum = 0.f;
    for (int s = sub; s < 512; s += 16) sum += __expf(sc[row * 520 + s] - m);
    for (int off = 8; off >= 1; off >>= 1) sum += __shfl_xor(sum, off, 16);
    float inv = 1.f / sum;
    for (int s = sub; s < 512; s += 16) {
      float p = __expf(sc[row * 520 + s] - m) * inv;
      ascore[(size_t)(b * T_ + t0 + row) * S_ + s] = p;
      pb[row * 520 + s] = (bhalf)p;
    }
  }
  __syncthreads();

  // ---- phase 3: attn_out = P @ mem ----
  const int dw = w * 64;
  v8f acc2[4] = {{}, {}, {}, {}};
  for (int sstep = 0; sstep < 16; ++sstep) {
    const int s0 = sstep * 32;
    __syncthreads();
    // stage mem[s0..s0+31][0..511] transposed into bt[d][k=s] (TDM cannot
    // transpose, so this stays a vector-load + scalar-store transpose)
    for (int cc = tid; cc < 2048; cc += 256) {
      int srow = cc >> 6, seg = (cc & 63) * 8;
      bhalf tmp[8];
      *(float4*)tmp =
          *(const float4*)(memb + (size_t)(b * S_ + s0 + srow) * D_ + seg);
      for (int e = 0; e < 8; ++e) bt[(seg + e) * 40 + srow] = tmp[e];
    }
    __syncthreads();
    v16bf a = frag_a(pb + s0, 520, lane);
    for (int f = 0; f < 4; ++f) {
      v16bf bb = frag_b(bt + (dw + f * 16) * 40, 40, lane);
      acc2[f] = wmma_bf16(a, bb, acc2[f]);
    }
  }
  {
    int roff = (lane & 16) ? 8 : 0, nc = lane & 15;
    for (int f = 0; f < 4; ++f) {
      int d = dw + f * 16 + nc;
      for (int r = 0; r < 8; ++r)
        attnb[(size_t)(b * T_ + t0 + roff + r) * D_ + d] = (bhalf)acc2[f][r];
    }
  }
}

// concat[row] = [dec(row) | attn(row)]
__global__ void k_concat(const bhalf* __restrict__ dec, const bhalf* __restrict__ attn,
                         bhalf* __restrict__ cat, int rows, int d) {
  int n = rows * 2 * d;
  for (int i = blockIdx.x * 256 + threadIdx.x; i < n; i += gridDim.x * 256) {
    int row = i / (2 * d), col = i % (2 * d);
    cat[i] = (col < d) ? dec[row * d + col] : attn[row * d + col - d];
  }
}

// p_gen = sigmoid(dec_attn @ Wg + bg), one wave per row
__global__ void k_pgen(const float* __restrict__ dattn, const float* __restrict__ Wg,
                       const float* __restrict__ bg, float* __restrict__ pgen,
                       int rows, int d) {
  int wave = (blockIdx.x * 256 + threadIdx.x) >> 5;
  int lane = threadIdx.x & 31;
  if (wave >= rows) return;
  const float* r = dattn + (size_t)wave * d;
  float s = 0.f;
  for (int i = lane; i < d; i += 32) s += r[i] * Wg[i];
  for (int off = 16; off >= 1; off >>= 1) s += __shfl_xor(s, off, 32);
  if (lane == 0) pgen[wave] = 1.f / (1.f + __expf(-(s + bg[0])));
}

// online (single pass) row max + sum(exp) over V
__global__ void k_rowstats(const float* __restrict__ logits, float* __restrict__ stats,
                           int V) {
  __shared__ float sm[256], ss[256];
  int row = blockIdx.x, tid = threadIdx.x;
  const float* p = logits + (size_t)row * V;
  float m = -INFINITY, s = 0.f;
  for (int v = tid; v < V; v += 256) {
    float x = p[v];
    float nm = fmaxf(m, x);
    s = s * __expf(m - nm) + __expf(x - nm);
    m = nm;
  }
  sm[tid] = m; ss[tid] = s;
  __syncthreads();
  for (int off = 128; off >= 1; off >>= 1) {
    if (tid < off) {
      float m2 = sm[tid + off], s2 = ss[tid + off];
      float nm = fmaxf(sm[tid], m2);
      ss[tid] = ss[tid] * __expf(sm[tid] - nm) + s2 * __expf(m2 - nm);
      sm[tid] = nm;
    }
    __syncthreads();
  }
  if (tid == 0) { stats[row * 2] = sm[0]; stats[row * 2 + 1] = ss[0]; }
}

// out = softmax(logits) * p_gen   (in place)
__global__ void k_mix(float* __restrict__ out, const float* __restrict__ stats,
                      const float* __restrict__ pgen, int V) {
  int row = blockIdx.y;
  int v = blockIdx.x * 256 + threadIdx.x;
  if (v >= V) return;
  size_t i = (size_t)row * V + v;
  out[i] = __expf(out[i] - stats[row * 2]) * (pgen[row] / stats[row * 2 + 1]);
}

// deterministic copy-prob scatter: one thread per (b,t) row (no float atomics,
// so replayed graphs produce bit-identical output)
__global__ void k_scatter(float* __restrict__ out, const float* __restrict__ ascore,
                          const float* __restrict__ pgen, const int* __restrict__ ids) {
  int r = blockIdx.x * 256 + threadIdx.x;
  if (r >= BT) return;
  int b = r / T_;
  float wgt = 1.f - pgen[r];
  float* o = out + (size_t)r * V_;
  const float* a = ascore + (size_t)r * S_;
  const int* id = ids + b * S_;
  for (int s = 0; s < S_; ++s) o[id[s]] += a[s] * wgt;
}

__global__ void k_log(float* __restrict__ out, int V) {
  int row = blockIdx.y;
  int v = blockIdx.x * 256 + threadIdx.x;
  if (v >= V) return;
  size_t i = (size_t)row * V + v;
  out[i] = logf(out[i]);
}

// ---------------------------------------------------------------------------
extern "C" void kernel_launch(void* const* d_in, const int* in_sizes, int n_in,
                              void* d_out, int out_size, void* d_ws, size_t ws_size,
                              hipStream_t stream) {
  const float* dec = (const float*)d_in[0];
  const float* mem = (const float*)d_in[1];
  const float* Wc  = (const float*)d_in[2];
  const float* bc  = (const float*)d_in[3];
  const float* Wd  = (const float*)d_in[4];
  const float* bd  = (const float*)d_in[5];
  const float* Wg  = (const float*)d_in[6];
  const float* bg  = (const float*)d_in[7];
  const float* Wo  = (const float*)d_in[8];
  const float* bo  = (const float*)d_in[9];
  const int* ids    = (const int*)d_in[10];
  const int* seqlen = (const int*)d_in[11];
  float* out = (float*)d_out;

  // workspace carve (~66 MB total, bf16 Wo dominates and stays L2-resident)
  char* w = (char*)d_ws;
  size_t off = 0;
  auto alloc = [&](size_t bytes) {
    void* p = w + off;
    off = (off + bytes + 255) & ~(size_t)255;
    return p;
  };
  bhalf* decb  = (bhalf*)alloc((size_t)BT * D_ * 2);
  bhalf* memb  = (bhalf*)alloc((size_t)BS * D_ * 2);
  bhalf* Wcb   = (bhalf*)alloc((size_t)D_ * D_ * 2);
  bhalf* Wdb   = (bhalf*)alloc((size_t)2 * D_ * D_ * 2);
  bhalf* Wob   = (bhalf*)alloc((size_t)D_ * V_ * 2);
  bhalf* projb = (bhalf*)alloc((size_t)BS * D_ * 2);
  bhalf* attnb = (bhalf*)alloc((size_t)BT * D_ * 2);
  bhalf* catb  = (bhalf*)alloc((size_t)BT * 2 * D_ * 2);
  float* ascore = (float*)alloc((size_t)BT * S_ * 4);
  float* dattn  = (float*)alloc((size_t)BT * D_ * 4);
  float* pgen   = (float*)alloc((size_t)BT * 4);
  float* stats  = (float*)alloc((size_t)BT * 2 * 4);

  // 1) bf16 casts
  k_cast<<<256, 256, 0, stream>>>(dec, decb, BT * D_);
  k_cast<<<256, 256, 0, stream>>>(mem, memb, BS * D_);
  k_cast<<<128, 256, 0, stream>>>(Wc, Wcb, D_ * D_);
  k_cast<<<256, 256, 0, stream>>>(Wd, Wdb, 2 * D_ * D_);
  k_cast<<<2048, 256, 0, stream>>>(Wo, Wob, D_ * V_);

  // 2) proj_mem = mem @ Wc + bc  (bf16 out)
  k_gemm<0><<<dim3(D_ / 128, BS / 64), 256, 0, stream>>>(
      memb, Wcb, bc, nullptr, projb, BS, D_, D_);

  // 3) fused attention (scores, masked softmax, attn_out)
  k_attn<<<B_ * (T_ / 16), 256, 0, stream>>>(decb, memb, projb, seqlen, ascore, attnb);

  // 4) dec_attn = tanh([dec|attn] @ Wd + bd), then p_gen
  k_concat<<<256, 256, 0, stream>>>(decb, attnb, catb, BT, D_);
  k_gemm<1><<<dim3(D_ / 128, BT / 64), 256, 0, stream>>>(
      catb, Wdb, bd, dattn, nullptr, BT, D_, 2 * D_);
  k_pgen<<<(BT * 32) / 256, 256, 0, stream>>>(dattn, Wg, bg, pgen, BT, D_);

  // 5) logits = dec @ Wo + bo  -> d_out (fp32)
  k_gemm<0><<<dim3((V_ + 127) / 128, BT / 64), 256, 0, stream>>>(
      decb, Wob, bo, out, nullptr, BT, V_, D_);

  // 6) softmax over V, mix with p_gen, scatter copy probs, log
  k_rowstats<<<BT, 256, 0, stream>>>(out, stats, V_);
  dim3 g2((V_ + 255) / 256, BT);
  k_mix<<<g2, 256, 0, stream>>>(out, stats, pgen, V_);
  k_scatter<<<(BT + 255) / 256, 256, 0, stream>>>(out, ascore, pgen, ids);
  k_log<<<g2, 256, 0, stream>>>(out, V_);
}
